// Decoder_71768903516444
// MI455X (gfx1250) — matrix operand order repaired and checked by reading, not compiled, not run on hardware
//
#include <hip/hip_runtime.h>

// ---------------------------------------------------------------------------
// Types for CDNA5 WMMA (wave32): v_wmma_f32_16x16x32_bf16
// ---------------------------------------------------------------------------
typedef __attribute__((ext_vector_type(16))) __bf16 bf16x16; // A/B frag: 8 VGPRs
typedef __attribute__((ext_vector_type(8)))  float  f32x8;   // C/D frag: 8 VGPRs
typedef __attribute__((ext_vector_type(4)))  int    i32x4;   // 16-byte chunk

#define WMMA_BF16(a, b, c) \
  __builtin_amdgcn_wmma_f32_16x16x32_bf16(false, (a), false, (b), (short)0, (c), false, false)

__device__ __forceinline__ unsigned short f2bf(float f) {
  unsigned int b = __float_as_uint(f);
  b = b + 0x7FFFu + ((b >> 16) & 1u);          // round-to-nearest-even
  return (unsigned short)(b >> 16);
}
__device__ __forceinline__ float sigmoidf_(float x) { return 1.0f / (1.0f + __expf(-x)); }
__device__ __forceinline__ float tanhf_(float x)    { return 2.0f / (1.0f + __expf(-2.0f * x)) - 1.0f; }

// A-fragment (16x32 bf16, ISA layout): row = lane&15, kbase = (lane>=16)?8:0,
// chunks at [kbase, kbase+8) and [kbase+16, kbase+24)  -> two 16B loads 32B apart.
__device__ __forceinline__ bf16x16 load_frag_split(const unsigned short* p) {
  bf16x16 f;
  ((i32x4*)&f)[0] = *(const i32x4*)(p);
  ((i32x4*)&f)[1] = *(const i32x4*)(p + 16);
  return f;
}
// B-fragment pre-packed per lane: 32 contiguous bytes.
__device__ __forceinline__ bf16x16 load_frag_contig(const unsigned int* p) {
  bf16x16 f;
  ((i32x4*)&f)[0] = *(const i32x4*)(p);
  ((i32x4*)&f)[1] = *(const i32x4*)(p + 4);
  return f;
}

// ---------------------------------------------------------------------------
// Kernel 1: embedding gather -> bf16, layout [T][N][16]
// ---------------------------------------------------------------------------
__global__ void embed_bf16(const int* __restrict__ x, const float* __restrict__ emb,
                           unsigned short* __restrict__ xe) {
  int gid = blockIdx.x * blockDim.x + threadIdx.x;
  if (gid >= 64 * 128 * 16) return;
  int k = gid & 15;
  int r = gid >> 4;          // t*64 + n
  int t = r >> 6, n = r & 63;
  int token = x[n * 128 + t];
  xe[gid] = f2bf(emb[token * 16 + k]);
}

// ---------------------------------------------------------------------------
// Kernel 2: pack W [NN][Ksrc] (torch layout) into WMMA B-fragment tiles of
// B = W^T (K x NN).  Tile (kt,nt): per lane 8 dwords (32B).  B 32x16 layout:
// lane 0-15: n=lane, K = kt*32 + 2v,2v+1 (v<4), +8 for v>=4 ; lane 16-31: n=lane-16, K += 8.
// ---------------------------------------------------------------------------
__global__ void pack_wT(const float* __restrict__ W, int Ksrc, int ntiles, int ktiles,
                        unsigned int* __restrict__ dst) {
  int gid = blockIdx.x * blockDim.x + threadIdx.x;
  int total = ktiles * ntiles * 32;
  if (gid >= total) return;
  int lane = gid & 31;
  int tile = gid >> 5;                // = kt*ntiles + nt
  int nt = tile % ntiles;
  int kt = tile / ntiles;
  int n  = nt * 16 + (lane & 15);
  int kb = kt * 32 + ((lane >> 4) ? 8 : 0);
#pragma unroll
  for (int v = 0; v < 8; ++v) {
    int k0 = kb + 2 * v + (v >= 4 ? 8 : 0), k1 = k0 + 1;
    float f0 = (k0 < Ksrc) ? W[n * Ksrc + k0] : 0.0f;
    float f1 = (k1 < Ksrc) ? W[n * Ksrc + k1] : 0.0f;
    dst[(tile * 32 + lane) * 8 + v] = (unsigned int)f2bf(f0) | ((unsigned int)f2bf(f1) << 16);
  }
}

// ---------------------------------------------------------------------------
// Kernel 3: recurrent 3-layer LSTM + folded MLP-1.  4 WGs x 16 batch rows.
// 8 waves; wave w owns hidden columns [w*32, w*32+32) for all 4 gates.
// c-state lives in registers (lane-exact WMMA C/D layout), h in LDS bf16.
// An opaque (always-0) offset laundered through an empty asm each timestep
// keeps the ~344 weight-fragment loads inside the t-loop (no LICM -> no RF
// overflow/spill), while keeping load provenance rooted at the kernel args
// so they lower to global_load_b128 (LOADcnt only), NOT flat_load_b128
// (which would couple to DScnt and occupy the LDS pipe used by A-frags).
// ---------------------------------------------------------------------------
#define LDH 272   // 256 + 32B pad: conflict-free ds_load_b128 across 16 rows
#define LDE 40

__global__ __launch_bounds__(256) void lstm_phase1(
    const unsigned short* __restrict__ xe,
    const float* __restrict__ h0, const float* __restrict__ c0,
    const float* __restrict__ bih1, const float* __restrict__ bhh1,
    const float* __restrict__ bih2, const float* __restrict__ bhh2,
    const float* __restrict__ bih3, const float* __restrict__ bhh3,
    const float* __restrict__ b1,
    const unsigned int* __restrict__ pWih1, const unsigned int* __restrict__ pWhh1,
    const unsigned int* __restrict__ pWih2, const unsigned int* __restrict__ pWhh2,
    const unsigned int* __restrict__ pWih3, const unsigned int* __restrict__ pWhh3,
    const unsigned int* __restrict__ pW1,
    unsigned short* __restrict__ act) {
  __shared__ __align__(16) unsigned short hb[3][16 * LDH];
  __shared__ __align__(16) unsigned short eb[16 * LDE];
  __shared__ float biasLds[3][1024];
  __shared__ float b1Lds[256];

  const int tid = threadIdx.x;
  const int lane = tid & 31;
  const int wv = tid >> 5;
  const int lanelo = lane & 15;
  const int hihalf = lane >> 4;
  const int bbase = blockIdx.x * 16;
  const f32x8 zf = {0, 0, 0, 0, 0, 0, 0, 0};

  for (int j = tid; j < 1024; j += 256) {
    biasLds[0][j] = bih1[j] + bhh1[j];
    biasLds[1][j] = bih2[j] + bhh2[j];
    biasLds[2][j] = bih3[j] + bhh3[j];
  }
  b1Lds[tid] = b1[tid];
  for (int j = tid; j < 16 * LDE; j += 256) eb[j] = 0;  // K-pad stays zero
  for (int j = tid; j < 16 * 256; j += 256) {
    int m = j >> 8, k = j & 255;
    unsigned short v = f2bf(h0[(bbase + m) * 256 + k]);
    hb[0][m * LDH + k] = v; hb[1][m * LDH + k] = v; hb[2][m * LDH + k] = v;
  }
  float creg[3][2][8];
#pragma unroll
  for (int u = 0; u < 2; ++u) {
    int hc = wv * 32 + u * 16 + lanelo;
#pragma unroll
    for (int v = 0; v < 8; ++v) {
      int m = v + hihalf * 8;
      float cv = c0[(bbase + m) * 256 + hc];
      creg[0][u][v] = cv; creg[1][u][v] = cv; creg[2][u][v] = cv;
    }
  }
  __syncthreads();

  unsigned int wOff = 0;  // opaque zero, redefined every timestep

  for (int t = 0; t < 128; ++t) {
    // Loop-carried opaque offset: defeats LICM of the weight-fragment loads
    // without breaking global-addrspace provenance of the kernel args.
    asm volatile("" : "+s"(wOff));
    const unsigned int* pIh[3] = {pWih1 + wOff, pWih2 + wOff, pWih3 + wOff};
    const unsigned int* pHh[3] = {pWhh1 + wOff, pWhh2 + wOff, pWhh3 + wOff};
    const unsigned int* pH1   = pW1 + wOff;

    { // stage this step's embedding slice (16x16 = 256 elems = 1/thread)
      int m = tid >> 4, k = tid & 15;
      eb[m * LDE + k] = xe[((t * 64) + bbase + m) * 16 + k];
    }
    __syncthreads();

#pragma unroll
    for (int cell = 0; cell < 3; ++cell) {
      const unsigned short* A1;
      int ld1, kt1;
      if (cell == 0) { A1 = eb; ld1 = LDE; kt1 = 1; }
      else           { A1 = &hb[cell - 1][0]; ld1 = LDH; kt1 = 8; }
      const unsigned short* A2 = &hb[cell][0];

      f32x8 acc[4][2];
#pragma unroll
      for (int q = 0; q < 4; ++q)
#pragma unroll
        for (int u = 0; u < 2; ++u) acc[q][u] = zf;

      for (int kt = 0; kt < kt1; ++kt) {  // gates += A1 @ Wih^T
        bf16x16 a = load_frag_split(A1 + lanelo * ld1 + kt * 32 + hihalf * 8);
#pragma unroll
        for (int q = 0; q < 4; ++q)
#pragma unroll
          for (int u = 0; u < 2; ++u) {
            int nt = q * 16 + wv * 2 + u;
            bf16x16 b = load_frag_contig(pIh[cell] + ((kt * 64 + nt) * 32 + lane) * 8);
            acc[q][u] = WMMA_BF16(a, b, acc[q][u]);
          }
      }
#pragma unroll
      for (int kt = 0; kt < 8; ++kt) {    // gates += h_self @ Whh^T
        bf16x16 a = load_frag_split(A2 + lanelo * LDH + kt * 32 + hihalf * 8);
#pragma unroll
        for (int q = 0; q < 4; ++q)
#pragma unroll
          for (int u = 0; u < 2; ++u) {
            int nt = q * 16 + wv * 2 + u;
            bf16x16 b = load_frag_contig(pHh[cell] + ((kt * 64 + nt) * 32 + lane) * 8);
            acc[q][u] = WMMA_BF16(a, b, acc[q][u]);
          }
      }
      // elementwise cell update: lane -> column hc, VGPR v -> row m
      float hval[2][8];
#pragma unroll
      for (int u = 0; u < 2; ++u) {
        int hc = wv * 32 + u * 16 + lanelo;
        float bi = biasLds[cell][hc];
        float bf_ = biasLds[cell][256 + hc];
        float bg = biasLds[cell][512 + hc];
        float bo = biasLds[cell][768 + hc];
#pragma unroll
        for (int v = 0; v < 8; ++v) {
          float iv = sigmoidf_(acc[0][u][v] + bi);
          float fv = sigmoidf_(acc[1][u][v] + bf_);
          float gv = tanhf_(acc[2][u][v] + bg);
          float ov = sigmoidf_(acc[3][u][v] + bo);
          float cn = fv * creg[cell][u][v] + iv * gv;
          creg[cell][u][v] = cn;
          hval[u][v] = ov * tanhf_(cn);
        }
      }
      __syncthreads();   // all reads of old h done before overwrite
      unsigned short* Hout = &hb[cell][0];
#pragma unroll
      for (int u = 0; u < 2; ++u) {
        int hc = wv * 32 + u * 16 + lanelo;
#pragma unroll
        for (int v = 0; v < 8; ++v) {
          int m = v + hihalf * 8;
          Hout[m * LDH + hc] = f2bf(hval[u][v]);
        }
      }
      __syncthreads();
    }

    { // folded MLP-1: act = relu(h3 @ W1^T + b1), stored at row n*T+t
      f32x8 acc2[2]; acc2[0] = zf; acc2[1] = zf;
#pragma unroll
      for (int kt = 0; kt < 8; ++kt) {
        bf16x16 a = load_frag_split(&hb[2][0] + lanelo * LDH + kt * 32 + hihalf * 8);
#pragma unroll
        for (int u = 0; u < 2; ++u) {
          int nt = wv * 2 + u;
          bf16x16 b = load_frag_contig(pH1 + ((kt * 16 + nt) * 32 + lane) * 8);
          acc2[u] = WMMA_BF16(a, b, acc2[u]);
        }
      }
#pragma unroll
      for (int u = 0; u < 2; ++u) {
        int col = wv * 32 + u * 16 + lanelo;
        float bb = b1Lds[col];
#pragma unroll
        for (int v = 0; v < 8; ++v) {
          int m = v + hihalf * 8;
          float a_ = acc2[u][v] + bb;
          a_ = a_ > 0.0f ? a_ : 0.0f;
          act[(((bbase + m) * 128) + t) * 256 + col] = f2bf(a_);
        }
      }
    }
  }
}

// ---------------------------------------------------------------------------
// Kernel 4: head GEMM  out[8192,20000] = act[8192,256] @ W2^T + b2.
// Wave task = 64(M) x 32(N) block: 4 Mtiles x 2 Ntiles, 8 f32x8 accumulators,
// B fragment reused across 4 Mtiles.  8192=128*64, 20000=625*32 exact.
// ---------------------------------------------------------------------------
__global__ __launch_bounds__(256) void head_gemm(
    const unsigned short* __restrict__ act, const unsigned int* __restrict__ pW2,
    const float* __restrict__ b2, float* __restrict__ out) {
  const int tid = threadIdx.x;
  const int lane = tid & 31, wv = tid >> 5;
  const int lanelo = lane & 15, hihalf = lane >> 4;
  const int task = blockIdx.x * 8 + wv;   // < 80000
  const int mg = task & 127;              // 128 M-groups of 4 tiles
  const int ng = task >> 7;               // 625 N-groups of 2 tiles
  const f32x8 zf = {0, 0, 0, 0, 0, 0, 0, 0};
  f32x8 acc[4][2];
#pragma unroll
  for (int i = 0; i < 4; ++i) { acc[i][0] = zf; acc[i][1] = zf; }

#pragma unroll
  for (int kt = 0; kt < 8; ++kt) {
    bf16x16 a[4];
#pragma unroll
    for (int i = 0; i < 4; ++i) {
      int row = (mg * 4 + i) * 16 + lanelo;
      a[i] = load_frag_split(act + row * 256 + kt * 32 + hihalf * 8);
    }
#pragma unroll
    for (int j = 0; j < 2; ++j) {
      int nt = ng * 2 + j;
      bf16x16 b = load_frag_contig(pW2 + ((kt * 1250 + nt) * 32 + lane) * 8);
#pragma unroll
      for (int i = 0; i < 4; ++i) acc[i][j] = WMMA_BF16(a[i], b, acc[i][j]);
    }
  }
#pragma unroll
  for (int j = 0; j < 2; ++j) {
    int n = (ng * 2 + j) * 16 + lanelo;
    float bb = b2[n];
#pragma unroll
    for (int i = 0; i < 4; ++i) {
      int m0 = (mg * 4 + i) * 16 + hihalf * 8;
#pragma unroll
      for (int v = 0; v < 8; ++v)
        out[(size_t)(m0 + v) * 20000 + n] = acc[i][j][v] + bb;
    }
  }
}

// Tail output: reference also returns x.reshape(-1)
__global__ void copy_x_tail(const int* __restrict__ x, float* __restrict__ outTail) {
  int gid = blockIdx.x * blockDim.x + threadIdx.x;
  if (gid < 8192) outTail[gid] = (float)x[gid];
}

// ---------------------------------------------------------------------------
extern "C" void kernel_launch(void* const* d_in, const int* in_sizes, int n_in,
                              void* d_out, int out_size, void* d_ws, size_t ws_size,
                              hipStream_t stream) {
  const int*   x    = (const int*)d_in[0];
  const float* h0   = (const float*)d_in[1];
  const float* c0   = (const float*)d_in[2];
  const float* emb  = (const float*)d_in[3];
  const float* Wih1 = (const float*)d_in[4];
  const float* Whh1 = (const float*)d_in[5];
  const float* bih1 = (const float*)d_in[6];
  const float* bhh1 = (const float*)d_in[7];
  const float* Wih2 = (const float*)d_in[8];
  const float* Whh2 = (const float*)d_in[9];
  const float* bih2 = (const float*)d_in[10];
  const float* bhh2 = (const float*)d_in[11];
  const float* Wih3 = (const float*)d_in[12];
  const float* Whh3 = (const float*)d_in[13];
  const float* bih3 = (const float*)d_in[14];
  const float* bhh3 = (const float*)d_in[15];
  const float* W1   = (const float*)d_in[16];
  const float* b1   = (const float*)d_in[17];
  const float* W2   = (const float*)d_in[18];
  const float* b2   = (const float*)d_in[19];

  char* ws = (char*)d_ws;
  constexpr size_t OFF_XE   = 0;               // 128*64*16*2      = 256 KB
  constexpr size_t OFF_ACT  = 262144;          // 8192*256*2       = 4 MB
  constexpr size_t OFF_WIH1 = 4456448;         // 1*64 tiles *1KB  = 64 KB
  constexpr size_t OFF_WHH1 = 4521984;         // 8*64*1KB         = 512 KB
  constexpr size_t OFF_WIH2 = 5046272;
  constexpr size_t OFF_WHH2 = 5570560;
  constexpr size_t OFF_WIH3 = 6094848;
  constexpr size_t OFF_WHH3 = 6619136;
  constexpr size_t OFF_W1   = 7143424;         // 8*16*1KB = 128 KB
  constexpr size_t OFF_W2   = 7274496;         // 8*1250*1KB = 10 MB (ends ~17.5 MB)

  unsigned short* xe  = (unsigned short*)(ws + OFF_XE);
  unsigned short* act = (unsigned short*)(ws + OFF_ACT);
  unsigned int* pWih1 = (unsigned int*)(ws + OFF_WIH1);
  unsigned int* pWhh1 = (unsigned int*)(ws + OFF_WHH1);
  unsigned int* pWih2 = (unsigned int*)(ws + OFF_WIH2);
  unsigned int* pWhh2 = (unsigned int*)(ws + OFF_WHH2);
  unsigned int* pWih3 = (unsigned int*)(ws + OFF_WIH3);
  unsigned int* pWhh3 = (unsigned int*)(ws + OFF_WHH3);
  unsigned int* pW1   = (unsigned int*)(ws + OFF_W1);
  unsigned int* pW2   = (unsigned int*)(ws + OFF_W2);

  embed_bf16<<<512, 256, 0, stream>>>(x, emb, xe);

  pack_wT<<<8,    256, 0, stream>>>(Wih1,  16, 64,   1, pWih1);
  pack_wT<<<64,   256, 0, stream>>>(Whh1, 256, 64,   8, pWhh1);
  pack_wT<<<64,   256, 0, stream>>>(Wih2, 256, 64,   8, pWih2);
  pack_wT<<<64,   256, 0, stream>>>(Whh2, 256, 64,   8, pWhh2);
  pack_wT<<<64,   256, 0, stream>>>(Wih3, 256, 64,   8, pWih3);
  pack_wT<<<64,   256, 0, stream>>>(Whh3, 256, 64,   8, pWhh3);
  pack_wT<<<16,   256, 0, stream>>>(W1,   256, 16,   8, pW1);
  pack_wT<<<1250, 256, 0, stream>>>(W2,   256, 1250, 8, pW2);

  lstm_phase1<<<4, 256, 0, stream>>>(xe, h0, c0, bih1, bhh1, bih2, bhh2, bih3, bhh3,
                                     b1, pWih1, pWhh1, pWih2, pWhh2, pWih3, pWhh3,
                                     pW1, act);

  head_gemm<<<10000, 256, 0, stream>>>(act, pW2, b2, (float*)d_out);
  copy_x_tail<<<32, 256, 0, stream>>>(x, (float*)d_out + (size_t)8192 * 20000);
}
